// Model_27650999452192
// MI455X (gfx1250) — compile-verified, matrix operand order checked by reading, not run on hardware
//
#include <hip/hip_runtime.h>
#include <hip/hip_bf16.h>

typedef __attribute__((ext_vector_type(16))) _Float16 v16h;
typedef __attribute__((ext_vector_type(8)))  float    v8f;
typedef __attribute__((ext_vector_type(4)))  float    v4f;
typedef int v4i_g __attribute__((vector_size(16)));   // GCC-style vector, matches builtin signature

#define H_DIM 1024
#define B_DIM 128
#define NWG   32      // persistent workgroups; each owns 32 hidden columns
#define TPB   256     // 8 wave32s; each wave owns a 16-row batch strip

union Frag16 { v16h v; v4f f[2]; };

// A fragment (16x32 f16): lane L<16 -> row M=L, K {0..7,16..23}; L>=16 -> row L-16, K {8..15,24..31}
__device__ __forceinline__ v16h load_a_frag(const _Float16* p, int hi8) {
  Frag16 u;
  u.f[0] = *(const v4f*)(p + hi8);
  u.f[1] = *(const v4f*)(p + 16 + hi8);
  return u.v;
}
// B fragment: packed so each lane reads 16 contiguous halfs (two b128 / ds_load_b128)
__device__ __forceinline__ v16h load_b_frag(const _Float16* p) {
  Frag16 u;
  u.f[0] = *(const v4f*)(p);
  u.f[1] = *(const v4f*)(p + 8);
  return u.v;
}

__device__ __forceinline__ float sigmf(float x) { return 1.f / (1.f + __expf(-x)); }

// ---- CDNA5 async global->LDS copy (ASYNCcnt-tracked), with guarded fallback ----
#if __has_builtin(__builtin_amdgcn_global_load_async_to_lds_b128)
#define HAVE_ASYNC_LDS 1
#endif

__device__ __forceinline__ void async_copy_b128(const _Float16* g, _Float16* l) {
#ifdef HAVE_ASYNC_LDS
  // flat->AS1 / flat->AS3 via integer round-trip (LDS flat addr low 32 bits == LDS offset).
  // Builtin signature (from hipcc diagnostic): (v4i AS1*, v4i AS3*, imm offset, imm cpol)
  __builtin_amdgcn_global_load_async_to_lds_b128(
      (__attribute__((address_space(1))) v4i_g*)(uintptr_t)g,
      (__attribute__((address_space(3))) v4i_g*)(unsigned)(uintptr_t)l,
      0, 0);
#else
  *(v4f*)l = *(const v4f*)g;   // staged through VGPRs: global_load_b128 + ds_store_b128
#endif
}

__device__ __forceinline__ void wait_async() {
#ifdef HAVE_ASYNC_LDS
#if __has_builtin(__builtin_amdgcn_s_wait_asynccnt)
  __builtin_amdgcn_s_wait_asynccnt(0);
#else
  asm volatile("s_wait_asynccnt 0x0" ::: "memory");
#endif
#endif
}

// Monotonic-counter grid barrier (all NWG workgroups are resident -> safe).
__device__ __forceinline__ void grid_sync(unsigned* cnt, unsigned target) {
  __threadfence();
  __syncthreads();
  if (threadIdx.x == 0) {
    atomicAdd(cnt, 1u);
    while (*((volatile unsigned*)cnt) < target)
      __builtin_amdgcn_s_sleep(8);
  }
  __syncthreads();
  __threadfence();
}

// MODE: 0 = layer0 with x input (F=8), 1 = layer0 with scalar y input, 2 = layer1 (K=2048, A=[h|h_lower])
template<int MODE>
__global__ void lstm_layer_kernel(
    const _Float16* __restrict__ Wp,      // packed weights, f16
    const float*    __restrict__ bias,    // combined bih+bhh [4096]
    const float*    __restrict__ in_f32,  // x (MODE 0) or y (MODE 1)
    int in_stride,
    const _Float16* __restrict__ hist_in, // lower-layer h history (MODE 2)
    const float*    __restrict__ wih,     // raw Wih (MODE 0: [4096][8], MODE 1: [4096][1])
    _Float16* __restrict__ hbuf0, _Float16* __restrict__ hbuf1,  // double-buffered h (f16)
    float*    __restrict__ cbuf,          // cell state f32 [B][H]
    _Float16* __restrict__ hist_out,      // h history out (may be null)
    unsigned* __restrict__ cnt, int T)
{
  constexpr int KT     = (MODE == 2) ? 64 : 32;   // K tiles of 32
  constexpr int CK     = 4;                       // kt-tiles per LDS chunk (32 KB)
  constexpr int NCHUNK = KT / CK;

  // Double-buffered B staging: all 8 waves share the same 8 N-tiles.
  __shared__ _Float16 ldsB[2][8][CK][512];        // 64 KB

  const int wg    = blockIdx.x;
  const int tid   = threadIdx.x;
  const int lane  = tid & 31;
  const int wave  = tid >> 5;
  const int m0    = wave * 16;                // batch-row base of this wave
  const int hi8   = (lane >> 4) << 3;         // lane-dependent K offset for A frags
  const int arow  = m0 + (lane & 15);
  const int col   = lane & 15;                // C/D: column within 16-tile
  const int rbase = (lane >> 4) << 3;         // C/D: rows 0-7 or 8-15

  // Cooperative async copy of one K-chunk of B into LDS (2048 x b128, 8 per thread).
  auto copy_chunk = [&](int c) {
    const int buf = c & 1;
    #pragma unroll
    for (int j = 0; j < 8; ++j) {
      const int idx = tid + j * TPB;          // 0..2047
      const int off = idx & 63;               // 16B unit inside one (nt,kt) 1KB region
      const int ktl = (idx >> 6) & (CK - 1);
      const int nt  = idx >> 8;               // 0..7
      const int ntg = (nt >> 1) * 64 + wg * 2 + (nt & 1);
      const _Float16* src = Wp + ((size_t)ntg * KT + (c * CK + ktl)) * 512 + off * 8;
      async_copy_b128(src, &ldsB[buf][nt][ktl][off * 8]);
    }
  };

  copy_chunk(0);
  wait_async();
  __syncthreads();

  for (int t = 0; t < T; ++t) {
    const _Float16* hsrc = (t & 1) ? hbuf1 : hbuf0;
    _Float16*       hdst = (t & 1) ? hbuf0 : hbuf1;

    const _Float16* aH = hsrc + (size_t)arow * H_DIM;
    const _Float16* aX = (MODE == 2) ? (hist_in + ((size_t)t * B_DIM + arow) * H_DIM) : nullptr;

    v8f zero8 = {0.f,0.f,0.f,0.f,0.f,0.f,0.f,0.f};
    v8f acc[8] = {zero8, zero8, zero8, zero8, zero8, zero8, zero8, zero8};

    for (int c = 0; c < NCHUNK; ++c) {
      if (c + 1 < NCHUNK) copy_chunk(c + 1);      // overlap next chunk copy with compute
      #pragma unroll
      for (int ktl = 0; ktl < CK; ++ktl) {
        const int kt = c * CK + ktl;
        const _Float16* ap = (MODE == 2 && kt >= 32) ? (aX + (kt - 32) * 32) : (aH + kt * 32);
        __builtin_prefetch(ap + 32, 0, 3);        // global_prefetch_b8 of next A frag
        v16h a = load_a_frag(ap, hi8);
        #pragma unroll
        for (int n = 0; n < 8; ++n) {             // gate gi = n>>1, sub-tile s = n&1
          v16h b = load_b_frag(&ldsB[c & 1][n][ktl][lane * 16]);
          acc[n] = __builtin_amdgcn_wmma_f32_16x16x32_f16(
              false, a, false, b, (short)0, acc[n], false, false);
        }
      }
      if (c + 1 == NCHUNK) copy_chunk(0);         // weights are time-invariant: preload next step
      wait_async();
      __syncthreads();
    }

    // Epilogue: this WG owns hidden cols j in [wg*32, wg*32+32); gates i,f,g,o are
    // acc[0+s], acc[2+s], acc[4+s], acc[6+s] for s in {0,1}.
    #pragma unroll
    for (int s = 0; s < 2; ++s) {
      const int j  = wg * 32 + s * 16 + col;
      const float bi = bias[j],            bf = bias[H_DIM + j];
      const float bg = bias[2*H_DIM + j],  bo = bias[3*H_DIM + j];
      float wi[8], wf[8], wgg[8], wo[8];
      float yi = 0.f, yf = 0.f, yg = 0.f, yo = 0.f;
      if (MODE == 0) {
        #pragma unroll
        for (int f = 0; f < 8; ++f) {
          wi[f]  = wih[((size_t)(0*H_DIM) + j) * 8 + f];
          wf[f]  = wih[((size_t)(1*H_DIM) + j) * 8 + f];
          wgg[f] = wih[((size_t)(2*H_DIM) + j) * 8 + f];
          wo[f]  = wih[((size_t)(3*H_DIM) + j) * 8 + f];
        }
      } else if (MODE == 1) {
        yi = wih[j]; yf = wih[H_DIM + j]; yg = wih[2*H_DIM + j]; yo = wih[3*H_DIM + j];
      }
      #pragma unroll
      for (int r = 0; r < 8; ++r) {
        const int b = m0 + rbase + r;
        float gi_v = acc[0 + s][r] + bi;
        float gf_v = acc[2 + s][r] + bf;
        float gg_v = acc[4 + s][r] + bg;
        float go_v = acc[6 + s][r] + bo;
        if (MODE == 0) {
          const float* xr = in_f32 + ((size_t)b * in_stride + t) * 8;
          #pragma unroll
          for (int f = 0; f < 8; ++f) {
            const float xv = xr[f];
            gi_v += xv * wi[f];  gf_v += xv * wf[f];
            gg_v += xv * wgg[f]; go_v += xv * wo[f];
          }
        } else if (MODE == 1) {
          const float yv = in_f32[(size_t)b * in_stride + t];
          gi_v += yv * yi; gf_v += yv * yf; gg_v += yv * yg; go_v += yv * yo;
        }
        const size_t idx = (size_t)b * H_DIM + j;
        const float c_old = cbuf[idx];
        const float cn = sigmf(gf_v) * c_old + sigmf(gi_v) * tanhf(gg_v);
        const float hn = sigmf(go_v) * tanhf(cn);
        cbuf[idx] = cn;
        hdst[idx] = (_Float16)hn;
        if (hist_out) hist_out[((size_t)t * B_DIM + b) * H_DIM + j] = (_Float16)hn;
      }
    }
    grid_sync(cnt, (unsigned)(t + 1) * NWG);
  }
}

// Pack W (row-major [4096][K]) into WMMA B-fragment order: 16 contiguous halfs per lane.
// idx -> (nt, kt, lane, e): element e of lane's v16h maps to K (e<8?e:e+8)+((lane>=16)?8:0).
__global__ void pack_w_kernel(const float* __restrict__ Whh, const float* __restrict__ Wih,
                              _Float16* __restrict__ out, int KT, int total) {
  int idx = blockIdx.x * blockDim.x + threadIdx.x;
  if (idx >= total) return;
  const int e    = idx & 15;
  const int lane = (idx >> 4) & 31;
  const int kt   = (idx >> 9) % KT;
  const int nt   = idx / (512 * KT);
  const int n    = nt * 16 + (lane & 15);
  const int kin  = (e < 8 ? e : e + 8) + ((lane >> 4) << 3);
  float v;
  if (kt < 32) v = Whh[(size_t)n * H_DIM + kt * 32 + kin];
  else         v = Wih[(size_t)n * H_DIM + (kt - 32) * 32 + kin];
  out[idx] = (_Float16)v;
}

__global__ void zero_kernel(float* p, int n) {
  int i = blockIdx.x * blockDim.x + threadIdx.x;
  if (i < n) p[i] = 0.f;
}

__global__ void bias_kernel(const float* a, const float* b, float* o) {
  int i = blockIdx.x * blockDim.x + threadIdx.x;
  if (i < 4 * H_DIM) o[i] = a[i] + b[i];
}

// mu/sigma heads + zero-fill of output_collection. One thread per (b,t).
__global__ void proj_kernel(const _Float16* __restrict__ hist,
                            const float* __restrict__ Wmu, const float* __restrict__ bmu,
                            const float* __restrict__ Wsig, const float* __restrict__ bsig,
                            float* __restrict__ out, int T) {
  int idx = blockIdx.x * blockDim.x + threadIdx.x;
  const int N = B_DIM * T;
  if (idx >= N) return;
  const int b = idx / T, t = idx % T;
  const _Float16* hp = hist + ((size_t)t * B_DIM + b) * H_DIM;
  float sm = 0.f, ss = 0.f;
  for (int j = 0; j < H_DIM; ++j) {
    const float hv = (float)hp[j];
    sm += hv * Wmu[j];
    ss += hv * Wsig[j];
  }
  sm += bmu[0];
  ss += bsig[0];
  const float sp = (ss > 20.f) ? ss : log1pf(__expf(ss));
  out[idx]         = 0.f;   // output_collection (zeros)
  out[N + idx]     = sm;    // mu
  out[2 * N + idx] = sp;    // sigma
}

extern "C" void kernel_launch(void* const* d_in, const int* in_sizes, int n_in,
                              void* d_out, int out_size, void* d_ws, size_t ws_size,
                              hipStream_t stream) {
  (void)in_sizes; (void)n_in; (void)out_size; (void)ws_size;
  const float* x        = (const float*)d_in[0];
  const float* y        = (const float*)d_in[1];
  const float* eWih0    = (const float*)d_in[2];
  const float* eWhh0    = (const float*)d_in[3];
  const float* ebih0    = (const float*)d_in[4];
  const float* ebhh0    = (const float*)d_in[5];
  const float* eWih1    = (const float*)d_in[6];
  const float* eWhh1    = (const float*)d_in[7];
  const float* ebih1    = (const float*)d_in[8];
  const float* ebhh1    = (const float*)d_in[9];
  const float* dWih0    = (const float*)d_in[10];
  const float* dWhh0    = (const float*)d_in[11];
  const float* dbih0    = (const float*)d_in[12];
  const float* dbhh0    = (const float*)d_in[13];
  const float* dWih1    = (const float*)d_in[14];
  const float* dWhh1    = (const float*)d_in[15];
  const float* dbih1    = (const float*)d_in[16];
  const float* dbhh1    = (const float*)d_in[17];
  const float* Wmu      = (const float*)d_in[18];
  const float* bmu      = (const float*)d_in[19];
  const float* Wsig     = (const float*)d_in[20];
  const float* bsig     = (const float*)d_in[21];

  char* ws = (char*)d_ws;
  size_t off = 0;
  auto alloc = [&](size_t bytes) { size_t o = off; off += (bytes + 255) & ~(size_t)255; return o; };

  // -------- zeroed state region (counters, biases, h, c) --------
  const size_t o_cnt  = alloc(256);
  const size_t o_be0  = alloc(4096 * 4), o_be1 = alloc(4096 * 4);
  const size_t o_bd0  = alloc(4096 * 4), o_bd1 = alloc(4096 * 4);
  const size_t o_h0l0 = alloc((size_t)B_DIM * H_DIM * 2), o_h1l0 = alloc((size_t)B_DIM * H_DIM * 2);
  const size_t o_h0l1 = alloc((size_t)B_DIM * H_DIM * 2), o_h1l1 = alloc((size_t)B_DIM * H_DIM * 2);
  const size_t o_cl0  = alloc((size_t)B_DIM * H_DIM * 4), o_cl1  = alloc((size_t)B_DIM * H_DIM * 4);
  const size_t zero_end = off;
  // -------- packed weights (f16) --------
  const size_t o_wpe0 = alloc((size_t)4096 * 1024 * 2);
  const size_t o_wpe1 = alloc((size_t)4096 * 2048 * 2);
  const size_t o_wpd0 = alloc((size_t)4096 * 1024 * 2);
  const size_t o_wpd1 = alloc((size_t)4096 * 2048 * 2);
  // -------- h histories (f16) --------
  const size_t o_he0  = alloc((size_t)256 * B_DIM * H_DIM * 2);
  const size_t o_hd0  = alloc((size_t)128 * B_DIM * H_DIM * 2);
  const size_t o_hd1  = alloc((size_t)128 * B_DIM * H_DIM * 2);

  unsigned*  cnt  = (unsigned*)(ws + o_cnt);
  float*     be0  = (float*)(ws + o_be0); float* be1 = (float*)(ws + o_be1);
  float*     bd0  = (float*)(ws + o_bd0); float* bd1 = (float*)(ws + o_bd1);
  _Float16*  h0l0 = (_Float16*)(ws + o_h0l0); _Float16* h1l0 = (_Float16*)(ws + o_h1l0);
  _Float16*  h0l1 = (_Float16*)(ws + o_h0l1); _Float16* h1l1 = (_Float16*)(ws + o_h1l1);
  float*     cl0  = (float*)(ws + o_cl0); float* cl1 = (float*)(ws + o_cl1);
  _Float16*  wpe0 = (_Float16*)(ws + o_wpe0); _Float16* wpe1 = (_Float16*)(ws + o_wpe1);
  _Float16*  wpd0 = (_Float16*)(ws + o_wpd0); _Float16* wpd1 = (_Float16*)(ws + o_wpd1);
  _Float16*  he0  = (_Float16*)(ws + o_he0);
  _Float16*  hd0  = (_Float16*)(ws + o_hd0);
  _Float16*  hd1  = (_Float16*)(ws + o_hd1);

  // init state (counters, h0, c) to zero each call -> deterministic
  const int nz = (int)(zero_end / 4);
  zero_kernel<<<(nz + 255) / 256, 256, 0, stream>>>((float*)ws, nz);

  bias_kernel<<<16, 256, 0, stream>>>(ebih0, ebhh0, be0);
  bias_kernel<<<16, 256, 0, stream>>>(ebih1, ebhh1, be1);
  bias_kernel<<<16, 256, 0, stream>>>(dbih0, dbhh0, bd0);
  bias_kernel<<<16, 256, 0, stream>>>(dbih1, dbhh1, bd1);

  const int tot0 = 4096 * 1024;   // KT=32 pack
  const int tot1 = 4096 * 2048;   // KT=64 pack ([Whh | Wih] fused along K)
  pack_w_kernel<<<tot0 / 256, 256, 0, stream>>>(eWhh0, nullptr, wpe0, 32, tot0);
  pack_w_kernel<<<tot1 / 256, 256, 0, stream>>>(eWhh1, eWih1,   wpe1, 64, tot1);
  pack_w_kernel<<<tot0 / 256, 256, 0, stream>>>(dWhh0, nullptr, wpd0, 32, tot0);
  pack_w_kernel<<<tot1 / 256, 256, 0, stream>>>(dWhh1, dWih1,   wpd1, 64, tot1);

  // Encoder (T=256), then decoder (T=128) continuing from encoder's final h/c.
  // T even => final h lands back in buffer 0, which is what the next kernel reads at t=0.
  lstm_layer_kernel<0><<<NWG, TPB, 0, stream>>>(wpe0, be0, x, 256, nullptr, eWih0,
                                                h0l0, h1l0, cl0, he0, cnt + 0, 256);
  lstm_layer_kernel<2><<<NWG, TPB, 0, stream>>>(wpe1, be1, nullptr, 0, he0, nullptr,
                                                h0l1, h1l1, cl1, nullptr, cnt + 1, 256);
  lstm_layer_kernel<1><<<NWG, TPB, 0, stream>>>(wpd0, bd0, y, 129, nullptr, dWih0,
                                                h0l0, h1l0, cl0, hd0, cnt + 2, 128);
  lstm_layer_kernel<2><<<NWG, TPB, 0, stream>>>(wpd1, bd1, nullptr, 0, hd0, nullptr,
                                                h0l1, h1l1, cl1, hd1, cnt + 3, 128);

  proj_kernel<<<(B_DIM * 128 + 255) / 256, 256, 0, stream>>>(hd1, Wmu, bmu, Wsig, bsig,
                                                             (float*)d_out, 128);
}
